// MetricRegressor_15659450761747
// MI455X (gfx1250) — matrix operand order repaired and checked by reading, not compiled, not run on hardware
//
#include <hip/hip_runtime.h>
#include <hip/hip_bf16.h>

// ---------------------------------------------------------------------------
// MetricRegressor for MI455X (gfx1250, wave32, WMMA).
//
//   sq_dist = |s|^2 + |c|^2 - 2 s@centers^T         [N,K]  (bf16 WMMA, f32 acc)
//   w       = |cw| * exp(-t * sq_dist)              (pointwise, f32)
//   mean    = (w @ means) / (rowsum(w) + 1)         [N,A]  (bf16 WMMA, f32 acc)
//   chol    = diag(exp(log_sigma))                  [A,A]
//
// N=65536, K=1024, D=256, A=64.  Compute bound (~460 FLOP/byte).
// Each 32-cluster chunk of centers (16 KB) and meansT (4 KB) is staged into
// LDS once per workgroup (double-buffered) via the CDNA5 async-to-LDS path
// (ASYNCcnt-tracked, overlapped with WMMA); all 8 waves share one copy.
// Fragments are ds_load_b128 from padded rows (bank-conflict-free).
// ---------------------------------------------------------------------------

typedef __attribute__((ext_vector_type(16))) __bf16 v16bf;
typedef __attribute__((ext_vector_type(8)))  __bf16 v8bf;
typedef __attribute__((ext_vector_type(8)))  float  v8f;
typedef __attribute__((ext_vector_type(4)))  int    v4i;

#define NN 65536
#define KK 1024
#define DD 256
#define AA 64

#ifndef USE_ASYNC_LDS
#if defined(__has_builtin)
#if __has_builtin(__builtin_amdgcn_global_load_async_to_lds_b128)
#define USE_ASYNC_LDS 1
#endif
#endif
#endif
#ifndef USE_ASYNC_LDS
#define USE_ASYNC_LDS 0
#endif

// Copy 16 bytes (8 bf16) global -> LDS.
static __device__ __forceinline__ void copy16_to_lds(__bf16* dst, const __bf16* src) {
#if USE_ASYNC_LDS
    __builtin_amdgcn_global_load_async_to_lds_b128(
        (__attribute__((address_space(1))) v4i*)(src),
        (__attribute__((address_space(3))) v4i*)(dst), 0, 0);
#else
    *(v8bf*)dst = *(const v8bf*)src;
#endif
}

static __device__ __forceinline__ void wait_async_lds() {
#if USE_ASYNC_LDS
#if defined(__has_builtin) && __has_builtin(__builtin_amdgcn_s_wait_asynccnt)
    __builtin_amdgcn_s_wait_asynccnt(0);
#else
    asm volatile("s_wait_asynccnt 0" ::: "memory");
#endif
#endif
}

// ---------------------------------------------------------------------------
// Prep: centers -> bf16, c2[k] = |centers_k|^2, cwabs[k] = |c_weights[k]|
// ---------------------------------------------------------------------------
__global__ void prep_centers_kernel(const float* __restrict__ centers,
                                    const float* __restrict__ c_weights,
                                    __bf16* __restrict__ centersB,
                                    float* __restrict__ c2,
                                    float* __restrict__ cwabs) {
    __shared__ float red[256];
    const int k   = blockIdx.x;
    const int tid = threadIdx.x;
    float v = centers[k * DD + tid];
    centersB[k * DD + tid] = (__bf16)v;
    red[tid] = v * v;
    __syncthreads();
    #pragma unroll
    for (int s = 128; s > 0; s >>= 1) {
        if (tid < s) red[tid] += red[tid + s];
        __syncthreads();
    }
    if (tid == 0) {
        c2[k]    = red[0];
        cwabs[k] = fabsf(c_weights[k]);
    }
}

// means [K,A] f32 -> meansT [A,K] bf16
__global__ void prep_meansT_kernel(const float* __restrict__ means,
                                   __bf16* __restrict__ meansT) {
    const int idx = blockIdx.x * 256 + threadIdx.x;   // a*1024 + k
    const int a = idx >> 10;
    const int k = idx & 1023;
    meansT[idx] = (__bf16)means[k * AA + a];
}

// chol = diag(exp(log_sigma))  [64 x 64]
__global__ void chol_kernel(const float* __restrict__ log_sigma,
                            float* __restrict__ outc) {
    const int idx = blockIdx.x * 256 + threadIdx.x;   // i*64 + j
    const int i = idx >> 6;
    const int j = idx & 63;
    outc[idx] = (i == j) ? __expf(log_sigma[i]) : 0.0f;
}

// ---------------------------------------------------------------------------
// Main kernel: 256 threads = 8 waves; each wave owns 16 rows of s and the full
// K loop; the workgroup cooperatively stages centers/meansT chunks in LDS.
//
// LDS row padding: centers rows 264 bf16 (528 B = 132 dwords; 132 mod 64 = 4)
// and meansT rows 40 bf16 (80 B = 20 dwords; gcd(20,64)=4): the 16 lanes of a
// fragment read at 4-dword-strided banks -> conflict-free b128 ds loads.
//
// GEMM 1 (transposed): dotT[cl,row] = centers(16x32) x sT(32x16)
//   A-frag: lane: cluster = l15; elem j -> d = c*32 + (j<8 ? 8h+j : 16+8h+j-8)
//   B-frag: lane: srow = l15;   elem j -> d = c*32 + (j<8 ? 8h+j : 16+8h+j-8)  (registers)
//   C layout: lane holds (M = r + 8h = cluster, N = l15 = row) -> all per-row
//   scalars (|s|^2, rowsum, final divide) are per-lane + one shfl_xor(16).
// GEMM 2: accT[a,row] += meansT(16x32) x w(32x16); w B-frag built from dotT's
//   C-layout with 16 shfl_xor(16) half swaps (layouts line up exactly).
// ---------------------------------------------------------------------------
__launch_bounds__(256)
__global__ void metric_regressor_main(const float* __restrict__ s,
                                      const __bf16* __restrict__ centersB,
                                      const float* __restrict__ c2,
                                      const float* __restrict__ cwabs,
                                      const __bf16* __restrict__ meansT,
                                      const float* __restrict__ log_temp,
                                      float* __restrict__ out) {
    __shared__ __align__(16) __bf16 cbuf[2][32][264];  // 33792 B
    __shared__ __align__(16) __bf16 mbuf[2][64][40];   // 10240 B

    const int tid   = threadIdx.x;
    const int lane  = tid & 31;
    const int half  = lane >> 4;
    const int l15   = lane & 15;
    const int wave  = tid >> 5;
    const int mBase = blockIdx.x * 128 + wave * 16;
    const int row   = mBase + l15;     // this lane's s-row (output row)

    const float t = __expf(log_temp[0]);

    // ---- cooperative stage of one 32-cluster chunk into LDS --------------
    auto prefetch = [&](int buf, int kc) {
        #pragma unroll
        for (int i = 0; i < 4; ++i) {               // centers: 1024 x 16 B
            const int id = tid + 256 * i;
            const int r  = id >> 5;
            const int pc = id & 31;
            copy16_to_lds(&cbuf[buf][r][pc * 8],
                          centersB + (size_t)(kc + r) * DD + pc * 8);
        }
        {                                            // meansT: 256 x 16 B
            const int r  = tid >> 2;
            const int pm = tid & 3;
            copy16_to_lds(&mbuf[buf][r][pm * 8],
                          meansT + (size_t)r * KK + kc + pm * 8);
        }
    };

    // ---- preload s fragments (B-layout, registers) + |s|^2 ---------------
    v16bf sfrag[8];
    float s2p = 0.0f;
    #pragma unroll
    for (int c = 0; c < 8; ++c) {
        const float* sp = s + (size_t)row * DD + c * 32 + 8 * half;
        #pragma unroll
        for (int q = 0; q < 4; ++q) {
            float4 v = ((const float4*)sp)[q < 2 ? q : q + 2];
            s2p += v.x * v.x + v.y * v.y + v.z * v.z + v.w * v.w;
            sfrag[c][q * 4 + 0] = (__bf16)v.x;
            sfrag[c][q * 4 + 1] = (__bf16)v.y;
            sfrag[c][q * 4 + 2] = (__bf16)v.z;
            sfrag[c][q * 4 + 3] = (__bf16)v.w;
        }
    }
    const float s2row = s2p + __shfl_xor(s2p, 16);

    v8f acc[4];
    #pragma unroll
    for (int at = 0; at < 4; ++at) acc[at] = {};
    float rs = 0.0f;

    prefetch(0, 0);
    int p = 0;

    #pragma unroll 1
    for (int kc = 0; kc < KK; kc += 32) {
        wait_async_lds();       // buf p copies complete (async path)
        __syncthreads();        // visible to / finished by all waves
        if (kc + 32 < KK) prefetch(p ^ 1, kc + 32);   // overlap next chunk

        // ---- GEMM 1: two 16-cluster dot tiles over D=256 (LDS A-frags) ---
        v8f dot0 = {};
        v8f dot1 = {};
        #pragma unroll
        for (int c = 0; c < 8; ++c) {
            const __bf16* p0 = &cbuf[p][l15][c * 32 + 8 * half];
            v8bf lo0 = *(const v8bf*)p0;
            v8bf hi0 = *(const v8bf*)(p0 + 16);
            v16bf a0;
            #pragma unroll
            for (int j = 0; j < 8; ++j) { a0[j] = lo0[j]; a0[j + 8] = hi0[j]; }
            dot0 = __builtin_amdgcn_wmma_f32_16x16x32_bf16(
                false, a0, false, sfrag[c], (short)0, dot0, false, false);

            const __bf16* p1 = &cbuf[p][16 + l15][c * 32 + 8 * half];
            v8bf lo1 = *(const v8bf*)p1;
            v8bf hi1 = *(const v8bf*)(p1 + 16);
            v16bf a1;
            #pragma unroll
            for (int j = 0; j < 8; ++j) { a1[j] = lo1[j]; a1[j + 8] = hi1[j]; }
            dot1 = __builtin_amdgcn_wmma_f32_16x16x32_bf16(
                false, a1, false, sfrag[c], (short)0, dot1, false, false);
        }

        // ---- pointwise: w = |cw| * exp(-t * dist), rowsum ----------------
        float w0[8], w1[8];
        #pragma unroll
        for (int r = 0; r < 8; ++r) {
            const int cl0 = kc + r + 8 * half;
            const float d0 = s2row + c2[cl0] - 2.0f * dot0[r];
            const float v0 = cwabs[cl0] * __expf(-t * d0);
            w0[r] = v0; rs += v0;

            const int cl1 = cl0 + 16;
            const float d1 = s2row + c2[cl1] - 2.0f * dot1[r];
            const float v1 = cwabs[cl1] * __expf(-t * d1);
            w1[r] = v1; rs += v1;
        }

        // ---- reshape w (C-layout) -> B-fragment (32 cl x 16 rows) --------
        v16bf wb;
        #pragma unroll
        for (int j = 0; j < 8; ++j) {
            const float x0 = __shfl_xor(w0[j], 16);
            const float x1 = __shfl_xor(w1[j], 16);
            wb[j]     = (__bf16)(half ? x1 : w0[j]);
            wb[j + 8] = (__bf16)(half ? w1[j] : x0);
        }

        // ---- GEMM 2: accT += meansT(16x32) x w(32x16), 4 a-tiles ---------
        #pragma unroll
        for (int at = 0; at < 4; ++at) {
            const __bf16* mp = &mbuf[p][at * 16 + l15][16 * half];
            v8bf lo = *(const v8bf*)mp;
            v8bf hi = *(const v8bf*)(mp + 8);
            v16bf ma;
            #pragma unroll
            for (int j = 0; j < 8; ++j) { ma[j] = lo[j]; ma[j + 8] = hi[j]; }
            acc[at] = __builtin_amdgcn_wmma_f32_16x16x32_bf16(
                false, ma, false, wb, (short)0, acc[at], false, false);
        }

        p ^= 1;
    }

    // ---- epilogue: divide by (rowsum + 1), store mean --------------------
    const float rsAll = rs + __shfl_xor(rs, 16);
    const float inv   = 1.0f / (rsAll + 1.0f);
    #pragma unroll
    for (int at = 0; at < 4; ++at) {
        #pragma unroll
        for (int r = 0; r < 8; ++r) {
            const int a = at * 16 + r + 8 * half;
            out[(size_t)row * AA + a] = acc[at][r] * inv;
        }
    }
}

// ---------------------------------------------------------------------------
// Host launcher. Inputs (setup_inputs order):
//   0: s [N,D] f32   1: centers [K,D] f32   2: c_weights [K] f32
//   3: means [K,A] f32   4: log_sigma [A] f32   5: log_temp [] f32
// Output: mean [N,A] f32 then chol [A,A] f32, concatenated flat.
// ---------------------------------------------------------------------------
extern "C" void kernel_launch(void* const* d_in, const int* in_sizes, int n_in,
                              void* d_out, int out_size, void* d_ws, size_t ws_size,
                              hipStream_t stream) {
    const float* s         = (const float*)d_in[0];
    const float* centers   = (const float*)d_in[1];
    const float* c_weights = (const float*)d_in[2];
    const float* means     = (const float*)d_in[3];
    const float* log_sigma = (const float*)d_in[4];
    const float* log_temp  = (const float*)d_in[5];
    float* out = (float*)d_out;

    // Workspace layout (total 648 KB):
    char* ws = (char*)d_ws;
    __bf16* centersB = (__bf16*)(ws);                  // K*D bf16 = 512 KB
    __bf16* meansT   = (__bf16*)(ws + 512 * 1024);     // A*K bf16 = 128 KB
    float*  c2       = (float*) (ws + 640 * 1024);     // K f32    =   4 KB
    float*  cwabs    = (float*) (ws + 644 * 1024);     // K f32    =   4 KB

    prep_centers_kernel<<<KK, 256, 0, stream>>>(centers, c_weights,
                                                centersB, c2, cwabs);
    prep_meansT_kernel<<<(AA * KK) / 256, 256, 0, stream>>>(means, meansT);

    metric_regressor_main<<<NN / 128, 256, 0, stream>>>(
        s, centersB, c2, cwabs, meansT, log_temp, out);

    chol_kernel<<<(AA * AA) / 256, 256, 0, stream>>>(
        log_sigma, out + (size_t)NN * AA);
}
